// AP_RecurrentModel_42683384987800
// MI455X (gfx1250) — matrix operand order repaired and checked by reading, not compiled, not run on hardware
//
#include <hip/hip_runtime.h>

#define T_DIM 256
#define B_DIM 128
#define I_DIM 512
#define H_DIM 512
#define G4H   2048   // 4*H

typedef __attribute__((ext_vector_type(16))) __bf16       v16bf;
typedef __attribute__((ext_vector_type(8)))  float        v8f;
typedef __attribute__((ext_vector_type(4)))  unsigned int u32x4;

__device__ __forceinline__ __bf16 to_bf16(float f) {
  union { float f; unsigned u; } a; a.f = f;
  unsigned r = a.u + 0x7FFFu + ((a.u >> 16) & 1u);   // round-to-nearest-even
  unsigned short hs = (unsigned short)(r >> 16);
  __bf16 b; __builtin_memcpy(&b, &hs, 2);
  return b;
}

__device__ __forceinline__ float sigmoidf_(float x) { return 1.0f / (1.0f + expf(-x)); }

// A fragment: 16x32 bf16 tile of row-major A (leading dim ld).  (ISA 7.12.2)
__device__ __forceinline__ v16bf load_a_frag(const __bf16* A, int row, int ld, int k0, int lane) {
  const __bf16* p = A + (long)row * ld + k0 + ((lane >> 4) << 3);
  union { v16bf v; u32x4 u[2]; } r;
  r.u[0] = *(const u32x4*)(p);
  r.u[1] = *(const u32x4*)(p + 16);
  return r.v;
}

// B fragment: 32x16 tile with B[k][n] = W[n][k], W row-major (computes A*W^T).
__device__ __forceinline__ v16bf load_b_frag(const __bf16* Wt, int n, int ld, int k0, int lane) {
  const __bf16* p = Wt + (long)n * ld + k0 + ((lane >> 4) << 3);
  union { v16bf v; u32x4 u[2]; } r;
  r.u[0] = *(const u32x4*)(p);
  r.u[1] = *(const u32x4*)(p + 16);
  return r.v;
}

// ---------------------------------------------------------------- init / cast
__global__ void init_cast(const float* __restrict__ x, const float* __restrict__ W,
                          const float* __restrict__ Wm,
                          __bf16* xb, __bf16* Wxb, __bf16* Whb, __bf16* Wmb,
                          float* c, float* colsum) {
  long i = (long)blockIdx.x * blockDim.x + threadIdx.x;
  long stride = (long)gridDim.x * blockDim.x;
  const long NX = (long)T_DIM * B_DIM * I_DIM;
  for (long t = i; t < NX; t += stride) xb[t] = to_bf16(x[t]);
  for (long t = i; t < (long)G4H * I_DIM; t += stride) {
    long row = t / I_DIM, k = t - row * I_DIM;
    Wxb[t] = to_bf16(W[row * (I_DIM + H_DIM) + k]);
  }
  for (long t = i; t < (long)G4H * H_DIM; t += stride) {
    long row = t / H_DIM, k = t - row * H_DIM;
    Whb[t] = to_bf16(W[row * (I_DIM + H_DIM) + I_DIM + k]);
  }
  for (long t = i; t < (long)H_DIM * I_DIM; t += stride) Wmb[t] = to_bf16(Wm[t]);
  for (long t = i; t < (long)B_DIM * H_DIM; t += stride) c[t] = 0.0f;
  for (long t = i; t < 3 * H_DIM; t += stride) colsum[t] = 0.0f;
}

// -------------------------------- big parallel GEMM: C = A*W^T + bias (pipelined)
// grid.x = M/16 row tiles, grid.y = N/256; block = 128 (4 waves, 64 cols each)
__global__ void gemm16(const __bf16* __restrict__ A, const __bf16* __restrict__ Wt,
                       const float* __restrict__ bias, float* __restrict__ C,
                       int K, int ldc) {
  int lane = threadIdx.x & 31, wave = threadIdx.x >> 5;
  int rowbase = blockIdx.x * 16;
  int colbase = blockIdx.y * 256 + wave * 64;
  v8f acc[4] = {};
  v16bf a_cur = load_a_frag(A, rowbase + (lane & 15), K, 0, lane);
  v16bf b_cur[4];
#pragma unroll
  for (int j = 0; j < 4; ++j)
    b_cur[j] = load_b_frag(Wt, colbase + 16 * j + (lane & 15), K, 0, lane);
  for (int k0 = 32; k0 < K; k0 += 32) {
    v16bf a_nxt = load_a_frag(A, rowbase + (lane & 15), K, k0, lane);
    v16bf b_nxt[4];
#pragma unroll
    for (int j = 0; j < 4; ++j)
      b_nxt[j] = load_b_frag(Wt, colbase + 16 * j + (lane & 15), K, k0, lane);
#pragma unroll
    for (int j = 0; j < 4; ++j)
      acc[j] = __builtin_amdgcn_wmma_f32_16x16x32_bf16(false, a_cur, false, b_cur[j],
                                                       (short)0, acc[j], false, false);
    a_cur = a_nxt;
#pragma unroll
    for (int j = 0; j < 4; ++j) b_cur[j] = b_nxt[j];
  }
#pragma unroll
  for (int j = 0; j < 4; ++j)
    acc[j] = __builtin_amdgcn_wmma_f32_16x16x32_bf16(false, a_cur, false, b_cur[j],
                                                     (short)0, acc[j], false, false);
  int rhalf = (lane >> 4) * 8;
#pragma unroll
  for (int j = 0; j < 4; ++j)
#pragma unroll
    for (int r = 0; r < 8; ++r) {
      int row = rowbase + rhalf + r;
      int col = colbase + 16 * j + (lane & 15);
      C[(long)row * ldc + col] = acc[j][r] + bias[col];
    }
}

// =================== fully fused recurrent step: one kernel per timestep =========
// grid = 8 (16-row blocks); block = 512 (16 waves).
// Phase A: finish step t-1 (cosine gates, out[t-1], new h -> LDS bf16)
// Phase B: gates = LDS-h @ Wh^T + xg_t (WMMA, double-buffered), LayerNorm, act -> LDS
// Phase C: cell update, colsum atomics, cosine partials
__global__ void lstm_step(const __bf16* __restrict__ Wh, const float* __restrict__ xg_t,
                          const float* __restrict__ gamma, const float* __restrict__ beta,
                          const float* __restrict__ im_prev, const float* __restrict__ im_cur,
                          float* __restrict__ c, float* __restrict__ hpre,
                          float* __restrict__ rowpart,
                          const float* __restrict__ cs_prev, float* __restrict__ cs_cur,
                          float* __restrict__ cs_zero,
                          float* __restrict__ out_prev, int t) {
  __shared__ float  sAct[16][G4H];          // 128 KB: activated gates for 16 rows
  __shared__ __bf16 sH[16][H_DIM];          // 16 KB: h_{t} (bf16) = WMMA A operand
  __shared__ float  lsum[16][16], lsq[16][16], lmean[4][16], lrstd[4][16];

  int lane = threadIdx.x & 31, w = threadIdx.x >> 5;   // wave = row (phases A/C)
  int rowbase = blockIdx.x * 16;

  // ---------------- Phase A: finalize previous step, build h in LDS ----------------
  if (t > 0) {
    int grow = rowbase + w;
    float s_imc = 0.0f, s_c2 = 0.0f;
#pragma unroll
    for (int q = 0; q < 16; ++q) {
      int idx = lane * 16 + q;
      float cs  = cs_prev[idx] * (1.0f / (float)B_DIM);
      float imv = im_prev[(long)grow * H_DIM + idx];
      s_imc += imv * cs; s_c2 += cs * cs;
    }
#pragma unroll
    for (int m = 16; m; m >>= 1) {
      s_imc += __shfl_xor(s_imc, m, 32);
      s_c2  += __shfl_xor(s_c2,  m, 32);
    }
    float im2 = rowpart[grow * 4 + 0];
    float imh = rowpart[grow * 4 + 1];
    float h2  = rowpart[grow * 4 + 2];
    float nim = fmaxf(sqrtf(im2), 1e-6f);
    float nc  = fmaxf(sqrtf(s_c2), 1e-6f);
    float nh  = fmaxf(sqrtf(h2), 1e-6f);
    float gic = sigmoidf_((s_imc / (nim * nc) + 1.0f) * 0.5f);
    float gco = sigmoidf_((imh  / (nim * nh) + 1.0f) * 0.5f);
    float gs = gic + gco;
#pragma unroll
    for (int q = 0; q < 16; ++q) {
      int idx = lane * 16 + q;
      long ci = (long)grow * H_DIM + idx;
      float hm = hpre[ci] * gs;
      out_prev[ci] = hm;
      sH[w][idx] = to_bf16(hm);
    }
  } else {
    for (int e = threadIdx.x; e < 16 * H_DIM; e += 512) ((__bf16*)sH)[e] = to_bf16(0.0f);
  }
  __syncthreads();

  // ---------------- Phase B: WMMA gates + LayerNorm + activation ----------------
  int gate = w >> 2;                        // waves 4g..4g+3 cover gate g
  int colbase = (w & 3) * 128;              // column offset within the gate
  const __bf16* hptr = (const __bf16*)sH;
  v8f acc[8] = {};
  v16bf a_cur = load_a_frag(hptr, (lane & 15), H_DIM, 0, lane);
  v16bf b_cur[8];
#pragma unroll
  for (int j = 0; j < 8; ++j)
    b_cur[j] = load_b_frag(Wh, gate * 512 + colbase + 16 * j + (lane & 15), H_DIM, 0, lane);
  for (int k0 = 32; k0 < H_DIM; k0 += 32) {
    v16bf a_nxt = load_a_frag(hptr, (lane & 15), H_DIM, k0, lane);
    v16bf b_nxt[8];
#pragma unroll
    for (int j = 0; j < 8; ++j)
      b_nxt[j] = load_b_frag(Wh, gate * 512 + colbase + 16 * j + (lane & 15), H_DIM, k0, lane);
#pragma unroll
    for (int j = 0; j < 8; ++j)
      acc[j] = __builtin_amdgcn_wmma_f32_16x16x32_bf16(false, a_cur, false, b_cur[j],
                                                       (short)0, acc[j], false, false);
    a_cur = a_nxt;
#pragma unroll
    for (int j = 0; j < 8; ++j) b_cur[j] = b_nxt[j];
  }
#pragma unroll
  for (int j = 0; j < 8; ++j)
    acc[j] = __builtin_amdgcn_wmma_f32_16x16x32_bf16(false, a_cur, false, b_cur[j],
                                                     (short)0, acc[j], false, false);

  int rhalf = (lane >> 4) * 8;
#pragma unroll
  for (int j = 0; j < 8; ++j)
#pragma unroll
    for (int r = 0; r < 8; ++r) {
      int row = rowbase + rhalf + r;
      int col = gate * 512 + colbase + 16 * j + (lane & 15);
      acc[j][r] += xg_t[(long)row * G4H + col];     // x-part incl. bias b
    }
  float psum[8], psq[8];
#pragma unroll
  for (int r = 0; r < 8; ++r) { psum[r] = 0.0f; psq[r] = 0.0f; }
#pragma unroll
  for (int j = 0; j < 8; ++j)
#pragma unroll
    for (int r = 0; r < 8; ++r) { float v = acc[j][r]; psum[r] += v; psq[r] += v * v; }
#pragma unroll
  for (int m = 1; m < 16; m <<= 1)
#pragma unroll
    for (int r = 0; r < 8; ++r) {
      psum[r] += __shfl_xor(psum[r], m, 32);
      psq[r]  += __shfl_xor(psq[r],  m, 32);
    }
  if ((lane & 15) == 0) {
#pragma unroll
    for (int r = 0; r < 8; ++r) { lsum[w][rhalf + r] = psum[r]; lsq[w][rhalf + r] = psq[r]; }
  }
  __syncthreads();
  if (threadIdx.x < 64) {
    int g = threadIdx.x >> 4, row = threadIdx.x & 15;
    float s = 0.0f, q = 0.0f;
#pragma unroll
    for (int u = 0; u < 4; ++u) { s += lsum[g * 4 + u][row]; q += lsq[g * 4 + u][row]; }
    float mean = s * (1.0f / 512.0f);
    float var  = q * (1.0f / 512.0f) - mean * mean;
    lmean[g][row] = mean;
    lrstd[g][row] = rsqrtf(var + 1e-5f);
  }
  __syncthreads();
#pragma unroll
  for (int j = 0; j < 8; ++j)
#pragma unroll
    for (int r = 0; r < 8; ++r) {
      int rl = rhalf + r;
      int cg = colbase + 16 * j + (lane & 15);
      float y = (acc[j][r] - lmean[gate][rl]) * lrstd[gate][rl]
                * gamma[gate * 512 + cg] + beta[gate * 512 + cg];
      sAct[rl][gate * 512 + cg] = (gate == 2) ? tanhf(y) : sigmoidf_(y);
    }
  __syncthreads();

  // ---------------- Phase C: cell update + cosine partials ----------------
  {
    int grow = rowbase + w;
    float s_im2 = 0.0f, s_imh = 0.0f, s_h2 = 0.0f;
#pragma unroll
    for (int q = 0; q < 16; ++q) {
      int idx = lane * 16 + q;
      float ig = sAct[w][idx], fg = sAct[w][512 + idx];
      float gg = sAct[w][1024 + idx], og = sAct[w][1536 + idx];
      long ci = (long)grow * H_DIM + idx;
      float cn = fg * c[ci] + ig * gg;
      c[ci] = cn;
      float h = og * tanhf(cn);
      hpre[ci] = h;
      atomicAdd(&cs_cur[idx], cn);
      float imv = im_cur[ci];
      s_im2 += imv * imv; s_imh += imv * h; s_h2 += h * h;
    }
#pragma unroll
    for (int m = 16; m; m >>= 1) {
      s_im2 += __shfl_xor(s_im2, m, 32);
      s_imh += __shfl_xor(s_imh, m, 32);
      s_h2  += __shfl_xor(s_h2,  m, 32);
    }
    if (lane == 0) {
      rowpart[grow * 4 + 0] = s_im2;
      rowpart[grow * 4 + 1] = s_imh;
      rowpart[grow * 4 + 2] = s_h2;
    }
  }
  // prepare the buffer step t+1 will accumulate into (3-way rotation => race-free)
  if (blockIdx.x == 0)
    for (int e = threadIdx.x; e < H_DIM; e += 512) cs_zero[e] = 0.0f;
}

// --------------------------------------------------------------- block reduce
__device__ __forceinline__ float block_sum(float v) {
  __shared__ float lds[8];
#pragma unroll
  for (int m = 16; m; m >>= 1) v += __shfl_xor(v, m, 32);
  int wave = threadIdx.x >> 5;
  if ((threadIdx.x & 31) == 0) lds[wave] = v;
  __syncthreads();
  float tot = 0.0f;
#pragma unroll
  for (int w = 0; w < 8; ++w) tot += lds[w];
  __syncthreads();
  return tot;
}

// ------------------- finalizer: emit out[T-1]; grid = B rows, block = 256
__global__ void step_final(const float* __restrict__ im_t, const float* __restrict__ hpre,
                           const float* __restrict__ colsum_t, const float* __restrict__ rowpart,
                           float* __restrict__ out_t) {
  int row = blockIdx.x, tid = threadIdx.x;
  float s_imc = 0.0f, s_c2 = 0.0f;
  for (int idx = tid; idx < H_DIM; idx += 256) {
    float cs = colsum_t[idx] * (1.0f / (float)B_DIM);
    float imv = im_t[(long)row * H_DIM + idx];
    s_imc += imv * cs; s_c2 += cs * cs;
  }
  s_imc = block_sum(s_imc);
  s_c2  = block_sum(s_c2);
  float im2 = rowpart[row * 4 + 0];
  float imh = rowpart[row * 4 + 1];
  float h2  = rowpart[row * 4 + 2];
  float nim = fmaxf(sqrtf(im2), 1e-6f);
  float nc  = fmaxf(sqrtf(s_c2), 1e-6f);
  float nh  = fmaxf(sqrtf(h2), 1e-6f);
  float gic = sigmoidf_((s_imc / (nim * nc) + 1.0f) * 0.5f);
  float gco = sigmoidf_((imh  / (nim * nh) + 1.0f) * 0.5f);
  float gs = gic + gco;
  for (int idx = tid; idx < H_DIM; idx += 256) {
    long ci = (long)row * H_DIM + idx;
    out_t[ci] = hpre[ci] * gs;
  }
}

// ------------------------------------------------------------------ launcher
extern "C" void kernel_launch(void* const* d_in, const int* in_sizes, int n_in,
                              void* d_out, int out_size, void* d_ws, size_t ws_size,
                              hipStream_t stream) {
  const float* x     = (const float*)d_in[0];   // [T,B,I]
  const float* W     = (const float*)d_in[1];   // [4H, I+H]
  const float* b     = (const float*)d_in[2];   // [4H]
  const float* Wm    = (const float*)d_in[3];   // [H, I]
  const float* bm    = (const float*)d_in[4];   // [H]
  const float* gamma = (const float*)d_in[5];   // [4,H]
  const float* beta  = (const float*)d_in[6];   // [4,H]
  float* out = (float*)d_out;                   // [T,B,H]

  char* p = (char*)d_ws;
  auto take = [&](size_t bytes) { char* q = p; p += (bytes + 255) & ~(size_t)255; return (void*)q; };
  __bf16* xb      = (__bf16*)take((size_t)T_DIM * B_DIM * I_DIM * 2);
  __bf16* Wxb     = (__bf16*)take((size_t)G4H * I_DIM * 2);
  __bf16* Whb     = (__bf16*)take((size_t)G4H * H_DIM * 2);
  __bf16* Wmb     = (__bf16*)take((size_t)H_DIM * I_DIM * 2);
  float*  xg      = (float*)take((size_t)T_DIM * B_DIM * G4H * 4);
  float*  im      = (float*)take((size_t)T_DIM * B_DIM * H_DIM * 4);
  float*  c       = (float*)take((size_t)B_DIM * H_DIM * 4);
  float*  hpre    = (float*)take((size_t)B_DIM * H_DIM * 4);
  float*  colsum  = (float*)take((size_t)3 * H_DIM * 4);
  float*  rowpart = (float*)take((size_t)B_DIM * 4 * 4);

  init_cast<<<4096, 256, 0, stream>>>(x, W, Wm, xb, Wxb, Whb, Wmb, c, colsum);

  // xg = x @ Wx^T + b  : M = 32768, N = 2048, K = 512
  gemm16<<<dim3((T_DIM * B_DIM) / 16, G4H / 256), 128, 0, stream>>>(xb, Wxb, b, xg, I_DIM, G4H);
  // im = x @ Wm^T + bm : M = 32768, N = 512, K = 512
  gemm16<<<dim3((T_DIM * B_DIM) / 16, H_DIM / 256), 128, 0, stream>>>(xb, Wmb, bm, im, I_DIM, H_DIM);

  const size_t BH = (size_t)B_DIM * H_DIM;
  for (int t = 0; t < T_DIM; ++t) {
    const float* xg_t    = xg + (size_t)t * B_DIM * G4H;
    const float* im_cur  = im + (size_t)t * BH;
    const float* im_prev = im + (size_t)(t > 0 ? t - 1 : 0) * BH;
    float* out_prev      = out + (size_t)(t > 0 ? t - 1 : 0) * BH;
    const float* cs_prev = colsum + (size_t)((t + 2) % 3) * H_DIM;
    float* cs_cur        = colsum + (size_t)(t % 3) * H_DIM;
    float* cs_zero       = colsum + (size_t)((t + 1) % 3) * H_DIM;
    lstm_step<<<B_DIM / 16, 512, 0, stream>>>(Whb, xg_t, gamma, beta, im_prev, im_cur,
                                              c, hpre, rowpart, cs_prev, cs_cur, cs_zero,
                                              out_prev, t);
  }
  step_final<<<B_DIM, 256, 0, stream>>>(im + (size_t)(T_DIM - 1) * BH, hpre,
                                        colsum + (size_t)((T_DIM - 1) % 3) * H_DIM,
                                        rowpart, out + (size_t)(T_DIM - 1) * BH);
}
